// GraphNeuralNetwork_58428735095623
// MI455X (gfx1250) — compile-verified, matrix operand order checked by reading, not run on hardware
//
#include <hip/hip_runtime.h>
#include <hip/hip_bf16.h>

// ---------------------------------------------------------------------------
// GAT (3 layers) + mean-pool + MLP for MI455X (gfx1250, wave32, WMMA)
// GEMMs via v_wmma_f32_16x16x32_f16; attention softmax via L2 atomics.
// ---------------------------------------------------------------------------

#define NNODES 10000
#define NEDGES 160000
#define HEADS  4
#define FDIM   128
#define CH     512          // HEADS*FDIM
#define NGRAPH 16
#define NEG_SLOPE 0.2f

typedef __attribute__((ext_vector_type(16))) _Float16 v16h;
typedef __attribute__((ext_vector_type(8)))  float    v8f;

// ----- float <-> orderable-uint for atomic segment-max -----
static __device__ __forceinline__ unsigned f2ord(float x) {
    unsigned u = __float_as_uint(x);
    return (u & 0x80000000u) ? ~u : (u | 0x80000000u);
}
static __device__ __forceinline__ float ord2f(unsigned k) {
    return __uint_as_float((k & 0x80000000u) ? (k & 0x7fffffffu) : ~k);
}

// ---------------------------------------------------------------------------
__global__ void k_zero_u32(unsigned* p, int n) {
    int i = blockIdx.x * blockDim.x + threadIdx.x;
    if (i < n) p[i] = 0u;
}

__global__ void k_f32_to_f16(const float* __restrict__ x, _Float16* __restrict__ y, int n) {
    int i = blockIdx.x * blockDim.x + threadIdx.x;
    if (i < n) y[i] = (_Float16)x[i];
}

// W [K, 512] f32  ->  Wt [512, K] f16 (transposed so B-fragment loads are contiguous in K)
__global__ void k_w_transpose_f16(const float* __restrict__ W, _Float16* __restrict__ Wt, int K) {
    int i = blockIdx.x * blockDim.x + threadIdx.x;
    if (i >= K * CH) return;
    int c = i / K, k = i - c * K;
    Wt[i] = (_Float16)W[(size_t)k * CH + c];
}

// ---------------------------------------------------------------------------
// GEMM: C[10000,512] = A[10000,K] * Wt[512,K]^T  (f16 in, f32 accumulate)
// grid = 625 blocks (16 rows each), 256 threads = 8 waves, wave w -> cols [w*64, w*64+64)
// A/B fragment: lane l: m/n = l&15, kh = l>>4; halves kh*8+{0..7} and 16+kh*8+{0..7}.
__global__ __launch_bounds__(256)
void k_gemm_wmma(const _Float16* __restrict__ A, const _Float16* __restrict__ Bt,
                 float* __restrict__ C, int K) {
    const int lane = threadIdx.x & 31;
    const int wave = threadIdx.x >> 5;      // 0..7
    const int rblk = blockIdx.x;            // 16-row block
    const int m    = lane & 15;
    const int kh   = lane >> 4;
    const int row  = rblk * 16 + m;

    v8f acc[4] = {};
    for (int k0 = 0; k0 < K; k0 += 32) {
        v16h a;
        const _Float16* ap = A + (size_t)row * K + k0 + kh * 8;
        ((float4*)&a)[0] = *(const float4*)(ap);
        ((float4*)&a)[1] = *(const float4*)(ap + 16);
#pragma unroll
        for (int t = 0; t < 4; t++) {
            const int col = wave * 64 + t * 16 + m;   // row of Wt = output column
            const _Float16* bp = Bt + (size_t)col * K + k0 + kh * 8;
            v16h b;
            ((float4*)&b)[0] = *(const float4*)(bp);
            ((float4*)&b)[1] = *(const float4*)(bp + 16);
            acc[t] = __builtin_amdgcn_wmma_f32_16x16x32_f16(
                false, a, false, b, (short)0, acc[t], false, false);
        }
    }
    // C/D layout: lane n = l&15; VGPR r -> M = r + 8*(l>>4)
#pragma unroll
    for (int t = 0; t < 4; t++) {
        const int colBase = wave * 64 + t * 16;
#pragma unroll
        for (int r = 0; r < 8; r++) {
            const int orow = rblk * 16 + r + 8 * kh;
            C[(size_t)orow * CH + colBase + m] = acc[t][r];
        }
    }
}

// ---------------------------------------------------------------------------
// Per-node attention coefficients: s_src[n,h] = sum_f h[n,h,f]*a_s[h,f] (and s_dst)
// block = 128 threads = 4 waves, one wave per head, one block per node.
__global__ __launch_bounds__(128)
void k_coeff(const float* __restrict__ Hb, const float* __restrict__ as,
             const float* __restrict__ ad, float* __restrict__ ssrc,
             float* __restrict__ sdst) {
    const int n = blockIdx.x;
    const int head = threadIdx.x >> 5;
    const int lane = threadIdx.x & 31;
    const float* hp  = Hb + (size_t)n * CH + head * FDIM;
    const float* asp = as + head * FDIM;
    const float* adp = ad + head * FDIM;
    float s0 = 0.f, s1 = 0.f;
#pragma unroll
    for (int j = 0; j < 4; j++) {
        float hv = hp[lane + 32 * j];
        s0 += hv * asp[lane + 32 * j];
        s1 += hv * adp[lane + 32 * j];
    }
#pragma unroll
    for (int off = 16; off > 0; off >>= 1) {
        s0 += __shfl_xor(s0, off, 32);
        s1 += __shfl_xor(s1, off, 32);
    }
    if (lane == 0) {
        ssrc[n * HEADS + head] = s0;
        sdst[n * HEADS + head] = s1;
    }
}

// ---------------------------------------------------------------------------
// Edge pass 1: e = leaky_relu(s_src[src]+s_dst[dst]); segment-max via ordered atomicMax.
__global__ void k_edge_logit(const int* __restrict__ ei, const float* __restrict__ ssrc,
                             const float* __restrict__ sdst, float* __restrict__ ee,
                             unsigned* __restrict__ mkeys) {
    const int E2 = NEDGES + NNODES;
    int idx = blockIdx.x * blockDim.x + threadIdx.x;
    if (idx >= E2 * HEADS) return;
    int eid = idx >> 2, h = idx & 3;
    int s, d;
    if (eid < NEDGES) { s = ei[eid]; d = ei[NEDGES + eid]; }
    else              { s = d = eid - NEDGES; }
    float v = ssrc[s * HEADS + h] + sdst[d * HEADS + h];
    v = (v > 0.f) ? v : NEG_SLOPE * v;
    ee[idx] = v;
    atomicMax(&mkeys[d * HEADS + h], f2ord(v));
}

// Edge pass 2: w = exp(e - m[dst]); denom[dst] += w.
__global__ void k_edge_exp(const int* __restrict__ ei, const float* __restrict__ ee,
                           const unsigned* __restrict__ mkeys, float* __restrict__ wbuf,
                           float* __restrict__ denom) {
    const int E2 = NEDGES + NNODES;
    int idx = blockIdx.x * blockDim.x + threadIdx.x;
    if (idx >= E2 * HEADS) return;
    int eid = idx >> 2, h = idx & 3;
    int d = (eid < NEDGES) ? ei[NEDGES + eid] : (eid - NEDGES);
    float m = ord2f(mkeys[d * HEADS + h]);
    float wv = __expf(ee[idx] - m);
    wbuf[idx] = wv;
    atomicAdd(&denom[d * HEADS + h], wv);
}

// Aggregation: O[dst,h,:] += (w/denom[dst,h]) * Hb[src,h,:]   (one wave per (edge,head))
__global__ __launch_bounds__(256)
void k_agg(const int* __restrict__ ei, const float* __restrict__ wbuf,
           const float* __restrict__ denom, const float* __restrict__ Hb,
           float* __restrict__ Ob) {
    const int E2 = NEDGES + NNODES;
    int g = (blockIdx.x * (int)blockDim.x + threadIdx.x) >> 5;
    int lane = threadIdx.x & 31;
    if (g >= E2 * HEADS) return;
    int eid = g >> 2, h = g & 3;
    int s, d;
    if (eid < NEDGES) { s = ei[eid]; d = ei[NEDGES + eid]; }
    else              { s = d = eid - NEDGES; }
    float alpha = wbuf[g] / denom[d * HEADS + h];
    float4 v = ((const float4*)(Hb + (size_t)s * CH + h * FDIM))[lane];
    float* op = Ob + (size_t)d * CH + h * FDIM + lane * 4;
    atomicAdd(op + 0, alpha * v.x);
    atomicAdd(op + 1, alpha * v.y);
    atomicAdd(op + 2, alpha * v.z);
    atomicAdd(op + 3, alpha * v.w);
}

// Layers 0/1 epilogue: relu(O + b) -> f16 activations for next GEMM.
__global__ void k_bias_relu_f16(const float* __restrict__ Ob, const float* __restrict__ b,
                                _Float16* __restrict__ Af16, int n) {
    int i = blockIdx.x * blockDim.x + threadIdx.x;
    if (i >= n) return;
    float v = Ob[i] + b[i & (CH - 1)];
    Af16[i] = (_Float16)fmaxf(v, 0.f);
}

// Layer 2 epilogue: head-mean + b2, then per-graph atomic mean-pool accumulation.
__global__ void k_head_mean_pool(const float* __restrict__ Ob, const float* __restrict__ b2,
                                 const int* __restrict__ batch, float* __restrict__ sums,
                                 float* __restrict__ cnt) {
    int i = blockIdx.x * blockDim.x + threadIdx.x;
    if (i >= NNODES * FDIM) return;
    int n = i >> 7, f = i & (FDIM - 1);
    const float* op = Ob + (size_t)n * CH;
    float v = 0.25f * (op[f] + op[FDIM + f] + op[2 * FDIM + f] + op[3 * FDIM + f]) + b2[f];
    int g = batch[n];
    atomicAdd(&sums[g * FDIM + f], v);
    if (f == 0) atomicAdd(&cnt[g], 1.0f);
}

__global__ void k_pool_norm(float* __restrict__ sums, const float* __restrict__ cnt) {
    int i = blockIdx.x * blockDim.x + threadIdx.x;
    if (i >= NGRAPH * FDIM) return;
    sums[i] /= fmaxf(cnt[i >> 7], 1.0f);
}

__global__ void k_mlp1(const float* __restrict__ gmean, const float* __restrict__ Wr1,
                       const float* __restrict__ br1, float* __restrict__ z) {
    int i = blockIdx.x * blockDim.x + threadIdx.x;
    if (i >= NGRAPH * FDIM) return;
    int g = i >> 7, j = i & (FDIM - 1);
    float acc = br1[j];
    for (int k = 0; k < FDIM; k++) acc += gmean[g * FDIM + k] * Wr1[k * FDIM + j];
    z[i] = fmaxf(acc, 0.f);
}

__global__ void k_mlp2(const float* __restrict__ z, const float* __restrict__ Wr2,
                       const float* __restrict__ br2, float* __restrict__ out) {
    int i = blockIdx.x * blockDim.x + threadIdx.x;
    if (i >= NGRAPH * 2 * FDIM) return;
    int g = i >> 8, j = i & (2 * FDIM - 1);
    float acc = br2[j];
    for (int k = 0; k < FDIM; k++) acc += z[g * FDIM + k] * Wr2[k * 2 * FDIM + j];
    out[i] = acc;
}

// ---------------------------------------------------------------------------
extern "C" void kernel_launch(void* const* d_in, const int* in_sizes, int n_in,
                              void* d_out, int out_size, void* d_ws, size_t ws_size,
                              hipStream_t stream) {
    (void)in_sizes; (void)n_in; (void)out_size; (void)ws_size;
    const int E2 = NEDGES + NNODES;
    const int DIN = 64;

    const float* x     = (const float*)d_in[0];
    const int*   ei    = (const int*)d_in[1];
    const int*   batch = (const int*)d_in[2];
    const float* W[3]  = {(const float*)d_in[3], (const float*)d_in[7],  (const float*)d_in[11]};
    const float* As[3] = {(const float*)d_in[4], (const float*)d_in[8],  (const float*)d_in[12]};
    const float* Ad[3] = {(const float*)d_in[5], (const float*)d_in[9],  (const float*)d_in[13]};
    const float* Bi[3] = {(const float*)d_in[6], (const float*)d_in[10], (const float*)d_in[14]};
    const float* Wr1 = (const float*)d_in[15];
    const float* br1 = (const float*)d_in[16];
    const float* Wr2 = (const float*)d_in[17];
    const float* br2 = (const float*)d_in[18];
    float* out = (float*)d_out;

    // ---- carve workspace ----
    char* p = (char*)d_ws;
    auto carve = [&](size_t bytes) -> char* {
        char* r = p;
        p += (bytes + 255) & ~(size_t)255;
        return r;
    };
    _Float16* Af16 = (_Float16*)carve((size_t)NNODES * CH * 2);   // f16 activations
    _Float16* Wt   = (_Float16*)carve((size_t)CH * CH * 2);       // transposed f16 weights
    float* Hb    = (float*)carve((size_t)NNODES * CH * 4);        // GEMM output h
    float* Ob    = (float*)carve((size_t)NNODES * CH * 4);        // aggregation output
    float* ssrc  = (float*)carve((size_t)NNODES * HEADS * 4);
    float* sdst  = (float*)carve((size_t)NNODES * HEADS * 4);
    unsigned* mk = (unsigned*)carve((size_t)NNODES * HEADS * 4);
    float* denom = (float*)carve((size_t)NNODES * HEADS * 4);
    float* ee    = (float*)carve((size_t)E2 * HEADS * 4);
    float* wbuf  = (float*)carve((size_t)E2 * HEADS * 4);
    float* sums  = (float*)carve((size_t)NGRAPH * FDIM * 4);
    float* cnt   = (float*)carve((size_t)NGRAPH * 4);
    float* z     = (float*)carve((size_t)NGRAPH * FDIM * 4);

    auto cdiv = [](int a, int b) { return (a + b - 1) / b; };

    // initial activation conversion: x [N,64] -> f16
    k_f32_to_f16<<<cdiv(NNODES * DIN, 256), 256, 0, stream>>>(x, Af16, NNODES * DIN);

    const int Kdims[3] = {DIN, CH, CH};
    for (int L = 0; L < 3; L++) {
        const int K = Kdims[L];
        k_w_transpose_f16<<<cdiv(K * CH, 256), 256, 0, stream>>>(W[L], Wt, K);
        k_gemm_wmma<<<NNODES / 16, 256, 0, stream>>>(Af16, Wt, Hb, K);
        k_coeff<<<NNODES, 128, 0, stream>>>(Hb, As[L], Ad[L], ssrc, sdst);
        k_zero_u32<<<cdiv(NNODES * HEADS, 256), 256, 0, stream>>>(mk, NNODES * HEADS);
        k_zero_u32<<<cdiv(NNODES * HEADS, 256), 256, 0, stream>>>((unsigned*)denom, NNODES * HEADS);
        k_zero_u32<<<cdiv(NNODES * CH, 256), 256, 0, stream>>>((unsigned*)Ob, NNODES * CH);
        k_edge_logit<<<cdiv(E2 * HEADS, 256), 256, 0, stream>>>(ei, ssrc, sdst, ee, mk);
        k_edge_exp<<<cdiv(E2 * HEADS, 256), 256, 0, stream>>>(ei, ee, mk, wbuf, denom);
        k_agg<<<cdiv(E2 * HEADS * 32, 256), 256, 0, stream>>>(ei, wbuf, denom, Hb, Ob);
        if (L < 2)
            k_bias_relu_f16<<<cdiv(NNODES * CH, 256), 256, 0, stream>>>(Ob, Bi[L], Af16, NNODES * CH);
    }

    // layer-2 epilogue: head mean + bias, global mean pool, MLP head
    k_zero_u32<<<cdiv(NGRAPH * FDIM, 256), 256, 0, stream>>>((unsigned*)sums, NGRAPH * FDIM);
    k_zero_u32<<<1, 64, 0, stream>>>((unsigned*)cnt, NGRAPH);
    k_head_mean_pool<<<cdiv(NNODES * FDIM, 256), 256, 0, stream>>>(Ob, Bi[2], batch, sums, cnt);
    k_pool_norm<<<cdiv(NGRAPH * FDIM, 256), 256, 0, stream>>>(sums, cnt);
    k_mlp1<<<cdiv(NGRAPH * FDIM, 256), 256, 0, stream>>>(sums, Wr1, br1, z);
    k_mlp2<<<cdiv(NGRAPH * 2 * FDIM, 256), 256, 0, stream>>>(z, Wr2, br2, out);
}
